// GraphConvNet_61735859913211
// MI455X (gfx1250) — matrix operand order repaired and checked by reading, not compiled, not run on hardware
//
#include <hip/hip_runtime.h>
#include <hip/hip_bf16.h>

typedef __bf16 bf16_t;
typedef __attribute__((ext_vector_type(16))) __bf16 v16bf;
typedef __attribute__((ext_vector_type(8)))  float  v8f;
typedef __attribute__((ext_vector_type(4)))  unsigned int v4u;

union AV16 { v16bf v; v4u u[2]; };

// ---- WMMA helpers --------------------------------------------------------
// A-matrix (16x32 bf16) register layout: lane l holds row (l&15); halves
// K = 8h..8h+7 and 8h+16..8h+23 with h = l>>4  -> two 16B contiguous loads.
__device__ __forceinline__ v16bf load_a_bf(const bf16_t* p, int h) {
  AV16 r;
  r.u[0] = *reinterpret_cast<const v4u*>(p + 8 * h);
  r.u[1] = *reinterpret_cast<const v4u*>(p + 8 * h + 16);
  return r.v;
}

__device__ __forceinline__ v16bf load_a_f32(const float* p, int h) {
  v16bf r;
#pragma unroll
  for (int j = 0; j < 8; ++j) {
    r[j]     = (bf16_t)p[8 * h + j];
    r[j + 8] = (bf16_t)p[8 * h + 16 + j];
  }
  return r;
}

// B tiles are pre-packed so each lane reads 16 contiguous bf16 (32B).
__device__ __forceinline__ v16bf load_b_tile(const bf16_t* packed, int tile) {
  int lane = (int)(threadIdx.x & 31);
  AV16 r;
  const v4u* p = reinterpret_cast<const v4u*>(packed + (size_t)tile * 512) + lane * 2;
  r.u[0] = p[0];
  r.u[1] = p[1];
  return r.v;
}

__device__ __forceinline__ void wmma_bf16(v8f& c, v16bf a, v16bf b) {
  c = __builtin_amdgcn_wmma_f32_16x16x32_bf16(false, a, false, b, (short)0, c,
                                              false, false);
}

__device__ __forceinline__ float gelu_f(float x) {
  const float k = 0.7978845608028654f;
  float t = tanhf(k * (x + 0.044715f * x * x * x));
  return 0.5f * x * (1.0f + t);
}

__device__ __forceinline__ void zero_acc4(v8f* a) {
#pragma unroll
  for (int i = 0; i < 4; ++i) {
    v8f z = {0.f, 0.f, 0.f, 0.f, 0.f, 0.f, 0.f, 0.f};
    a[i] = z;
  }
}

// C layout: lane l holds col (l&15)+16*nt ; VGPR v holds row v + 8*(l>>4).
__device__ __forceinline__ void act_store_lds(const v8f* acc, const float* bias,
                                              bool act, bf16_t* tile /*16x72*/) {
  int lane = (int)(threadIdx.x & 31);
  int col0 = lane & 15, h = lane >> 4;
#pragma unroll
  for (int nt = 0; nt < 4; ++nt) {
    int col = nt * 16 + col0;
    float b = bias[col];
#pragma unroll
    for (int v = 0; v < 8; ++v) {
      float x = acc[nt][v] + b;
      if (act) x = gelu_f(x);
      tile[(v + 8 * h) * 72 + col] = (bf16_t)x;
    }
  }
}

// 64-wide hidden layer from an LDS tile (2 K-tiles).
__device__ __forceinline__ void layer64(const bf16_t* tile, const bf16_t* Wp,
                                        int ntiles, v8f* acc) {
  int lane = (int)(threadIdx.x & 31);
  int row = lane & 15, h = lane >> 4;
  for (int kt = 0; kt < 2; ++kt) {
    v16bf a = load_a_bf(tile + row * 72 + kt * 32, h);
    for (int nt = 0; nt < ntiles; ++nt)
      wmma_bf16(acc[nt], a, load_b_tile(Wp, kt * ntiles + nt));
  }
}

// ---- weight packing: W[K x Ncols] row-major -> B-register layout tiles ----
__global__ void pack_b_kernel(const float* W, int Krows, int Ncols, int ntiles,
                              bf16_t* dst) {
  int tile = blockIdx.x;
  int kt = tile / ntiles, nt = tile % ntiles;
  int lane = (int)threadIdx.x;
  int col = nt * 16 + (lane & 15);
  int h = lane >> 4;
  bf16_t* out = dst + (size_t)tile * 512 + lane * 16;
#pragma unroll
  for (int j = 0; j < 16; ++j) {
    int k = kt * 32 + 8 * h + j + (j >= 8 ? 8 : 0);
    float v = (k < Krows && col < Ncols) ? W[(size_t)k * Ncols + col] : 0.f;
    out[j] = (bf16_t)v;
  }
}

__global__ void pad_rows_kernel(const float* src, int R, int C, int Cpad,
                                bf16_t* dst) {
  long long i = (long long)blockIdx.x * blockDim.x + threadIdx.x;
  if (i >= (long long)R * Cpad) return;
  int r = (int)(i / Cpad), c = (int)(i % Cpad);
  dst[i] = (bf16_t)((c < C) ? src[(size_t)r * C + c] : 0.f);
}

__global__ void zero_f32_kernel(float* p, int n) {
  int i = blockIdx.x * blockDim.x + threadIdx.x;
  if (i < n) p[i] = 0.f;
}

// ---- generic fused 3-layer MLP on a dense bf16 matrix --------------------
// Used for node/edge embedders and the decoder.
__global__ void mlp3_dense_kernel(const bf16_t* X, int xStride, int kt1,
                                  const bf16_t* W0p, const float* b0,
                                  const bf16_t* W1p, const float* b1,
                                  const bf16_t* W2p, const float* b2, int n2t,
                                  float* outF, int outStride, int outCols,
                                  bf16_t* outB, int R) {
  __shared__ __align__(16) bf16_t lds[4][16 * 72];
  int wave = (int)(threadIdx.x >> 5);
  int lane = (int)(threadIdx.x & 31);
  int Rtiles = (R + 15) / 16;
  int tile = blockIdx.x * 4 + wave;
  bool active = tile < Rtiles;
  int t = active ? tile : 0;
  int rowb = t * 16;
  int row = lane & 15, h = lane >> 4;
  int grow = rowb + row;
  if (grow > R - 1) grow = R - 1;
  const bf16_t* xrow = X + (size_t)grow * xStride;

  v8f acc[4];
  zero_acc4(acc);
  for (int kt = 0; kt < kt1; ++kt) {
    v16bf a = load_a_bf(xrow + kt * 32, h);
    for (int nt = 0; nt < 4; ++nt)
      wmma_bf16(acc[nt], a, load_b_tile(W0p, kt * 4 + nt));
  }
  act_store_lds(acc, b0, true, lds[wave]);
  __syncthreads();

  zero_acc4(acc);
  layer64(lds[wave], W1p, 4, acc);
  __syncthreads();
  act_store_lds(acc, b1, true, lds[wave]);
  __syncthreads();

  zero_acc4(acc);
  layer64(lds[wave], W2p, n2t, acc);

  if (active) {
    int col0 = lane & 15;
#pragma unroll
    for (int v = 0; v < 8; ++v) {
      int r = rowb + v + 8 * h;
      if (r < R) {
        for (int nt = 0; nt < n2t; ++nt) {
          int col = nt * 16 + col0;
          if (col < outCols) {
            float x = acc[nt][v] + b2[col];
            if (outF) outF[(size_t)r * outStride + col] = x;
            if (outB) outB[(size_t)r * 64 + col] = (bf16_t)x;
          }
        }
      }
    }
  }
}

// ---- per-step edge update: gather-concat MLP + fused segment_sum ---------
__global__ void edge_step_kernel(const bf16_t* e, const bf16_t* n,
                                 const int* senders, const int* receivers,
                                 const bf16_t* glob_pad, const bf16_t* W0p,
                                 const float* b0, const bf16_t* W1p,
                                 const float* b1, const bf16_t* W2p,
                                 const float* b2, float* new_e, float* recv,
                                 int E) {
  __shared__ __align__(16) bf16_t lds[4][16 * 72];
  int wave = (int)(threadIdx.x >> 5);
  int lane = (int)(threadIdx.x & 31);
  int tile = blockIdx.x * 4 + wave;  // E/16 tiles exactly
  int rowb = tile * 16;
  int row = lane & 15, h = lane >> 4;
  int er = rowb + row;
  int snd = senders[er];
  int rcv = receivers[er];
  const bf16_t* p0 = e + (size_t)er * 64;
  const bf16_t* p1 = n + (size_t)snd * 64;
  const bf16_t* p2 = n + (size_t)rcv * 64;

  v8f acc[4];
  zero_acc4(acc);
  // X = [e | n_snd | n_rcv | glob,pad] : 7 K-tiles of 32
  for (int kt = 0; kt < 7; ++kt) {
    const bf16_t* p;
    if (kt < 2)      p = p0 + kt * 32;
    else if (kt < 4) p = p1 + (kt - 2) * 32;
    else if (kt < 6) p = p2 + (kt - 4) * 32;
    else             p = glob_pad;
    v16bf a = load_a_bf(p, h);
    for (int nt = 0; nt < 4; ++nt)
      wmma_bf16(acc[nt], a, load_b_tile(W0p, kt * 4 + nt));
  }
  act_store_lds(acc, b0, true, lds[wave]);
  __syncthreads();

  zero_acc4(acc);
  layer64(lds[wave], W1p, 4, acc);
  __syncthreads();
  act_store_lds(acc, b1, true, lds[wave]);
  __syncthreads();

  zero_acc4(acc);
  layer64(lds[wave], W2p, 4, acc);

  int col0 = lane & 15;
#pragma unroll
  for (int v = 0; v < 8; ++v) {
    int r = rowb + v + 8 * h;
    int rr = receivers[r];
#pragma unroll
    for (int nt = 0; nt < 4; ++nt) {
      int col = nt * 16 + col0;
      float x = acc[nt][v] + b2[col];
      new_e[(size_t)r * 64 + col] = x;
      atomicAdd(recv + (size_t)rr * 64 + col, x);  // segment_sum (L2-resident)
    }
  }
}

// ---- per-step node update -----------------------------------------------
__global__ void node_step_kernel(const bf16_t* n, const float* recv,
                                 const bf16_t* glob_pad, const bf16_t* W0p,
                                 const float* b0, const bf16_t* W1p,
                                 const float* b1, const bf16_t* W2p,
                                 const float* b2, float* new_n, int R) {
  __shared__ __align__(16) bf16_t lds[4][16 * 72];
  int wave = (int)(threadIdx.x >> 5);
  int lane = (int)(threadIdx.x & 31);
  int Rtiles = R / 16;  // R multiple of 16
  int tile = blockIdx.x * 4 + wave;
  bool active = tile < Rtiles;
  int t = active ? tile : 0;
  int rowb = t * 16;
  int row = lane & 15, h = lane >> 4;
  int nr = rowb + row;
  const bf16_t* pn = n + (size_t)nr * 64;
  const float* pr = recv + (size_t)nr * 64;

  v8f acc[4];
  zero_acc4(acc);
  // X = [n | recv | glob,pad] : 5 K-tiles of 32
  for (int kt = 0; kt < 5; ++kt) {
    v16bf a;
    if (kt < 2)      a = load_a_bf(pn + kt * 32, h);
    else if (kt < 4) a = load_a_f32(pr + (kt - 2) * 32, h);
    else             a = load_a_bf(glob_pad, h);
    for (int nt = 0; nt < 4; ++nt)
      wmma_bf16(acc[nt], a, load_b_tile(W0p, kt * 4 + nt));
  }
  act_store_lds(acc, b0, true, lds[wave]);
  __syncthreads();

  zero_acc4(acc);
  layer64(lds[wave], W1p, 4, acc);
  __syncthreads();
  act_store_lds(acc, b1, true, lds[wave]);
  __syncthreads();

  zero_acc4(acc);
  layer64(lds[wave], W2p, 4, acc);

  if (active) {
    int col0 = lane & 15;
#pragma unroll
    for (int v = 0; v < 8; ++v) {
      int r = rowb + v + 8 * h;
#pragma unroll
      for (int nt = 0; nt < 4; ++nt) {
        int col = nt * 16 + col0;
        new_n[(size_t)r * 64 + col] = acc[nt][v] + b2[col];
      }
    }
  }
}

// ---- residual + LayerNorm (wave per row, shfl_xor reduction) -------------
__global__ void ln_node_kernel(float* n, bf16_t* n_bf, const float* new_n,
                               const float* scale, const float* bias, int R) {
  int rrow = blockIdx.x * (blockDim.x >> 5) + (threadIdx.x >> 5);
  if (rrow >= R) return;
  int lane = (int)(threadIdx.x & 31);
  size_t base = (size_t)rrow * 64;
  float x0 = n[base + lane] + new_n[base + lane];
  float x1 = n[base + lane + 32] + new_n[base + lane + 32];
  float s = x0 + x1;
#pragma unroll
  for (int m = 16; m >= 1; m >>= 1) s += __shfl_xor(s, m, 32);
  float mu = s * (1.0f / 64.0f);
  float c0 = x0 - mu, c1 = x1 - mu;
  float vv = c0 * c0 + c1 * c1;
#pragma unroll
  for (int m = 16; m >= 1; m >>= 1) vv += __shfl_xor(vv, m, 32);
  float inv = rsqrtf(vv * (1.0f / 64.0f) + 1e-6f);
  float y0 = c0 * inv * scale[lane] + bias[lane];
  float y1 = c1 * inv * scale[lane + 32] + bias[lane + 32];
  n[base + lane] = y0;
  n[base + lane + 32] = y1;
  n_bf[base + lane] = (bf16_t)y0;
  n_bf[base + lane + 32] = (bf16_t)y1;
}

__global__ void ln_edge_kernel(bf16_t* e, const float* new_e,
                               const float* scale, const float* bias, int R) {
  int rrow = blockIdx.x * (blockDim.x >> 5) + (threadIdx.x >> 5);
  if (rrow >= R) return;
  int lane = (int)(threadIdx.x & 31);
  size_t base = (size_t)rrow * 64;
  float x0 = (float)e[base + lane] + new_e[base + lane];
  float x1 = (float)e[base + lane + 32] + new_e[base + lane + 32];
  float s = x0 + x1;
#pragma unroll
  for (int m = 16; m >= 1; m >>= 1) s += __shfl_xor(s, m, 32);
  float mu = s * (1.0f / 64.0f);
  float c0 = x0 - mu, c1 = x1 - mu;
  float vv = c0 * c0 + c1 * c1;
#pragma unroll
  for (int m = 16; m >= 1; m >>= 1) vv += __shfl_xor(vv, m, 32);
  float inv = rsqrtf(vv * (1.0f / 64.0f) + 1e-6f);
  e[base + lane] = (bf16_t)(c0 * inv * scale[lane] + bias[lane]);
  e[base + lane + 32] = (bf16_t)(c1 * inv * scale[lane + 32] + bias[lane + 32]);
}

// ---- host orchestration --------------------------------------------------
extern "C" void kernel_launch(void* const* d_in, const int* in_sizes, int n_in,
                              void* d_out, int out_size, void* d_ws,
                              size_t ws_size, hipStream_t stream) {
  (void)in_sizes; (void)n_in; (void)out_size; (void)ws_size;
  const int N = 50000, E = 800000, S = 3;

  const float* glob = (const float*)d_in[2];
  const int* senders = (const int*)d_in[37];
  const int* receivers = (const int*)d_in[38];

  char* ws = (char*)d_ws;
  size_t off = 0;
  auto take = [&](size_t bytes) -> char* {
    char* p = ws + off;
    off = (off + bytes + 255) & ~(size_t)255;
    return p;
  };
  auto takeT = [&](int tiles) -> bf16_t* {
    return (bf16_t*)take((size_t)tiles * 512 * sizeof(bf16_t));
  };

  float*  n_f32  = (float*)take((size_t)N * 64 * 4);
  bf16_t* n_bf16 = (bf16_t*)take((size_t)N * 64 * 2);
  float*  new_n  = (float*)take((size_t)N * 64 * 4);
  float*  recv   = (float*)take((size_t)N * 64 * 4);
  bf16_t* e_bf16 = (bf16_t*)take((size_t)E * 64 * 2);
  float*  new_e  = (float*)take((size_t)E * 64 * 4);
  bf16_t* Xn = (bf16_t*)new_n;  // alias: padded raw nodes, dead before step 0
  bf16_t* Xe = (bf16_t*)new_e;  // alias: padded raw edges, dead before step 0
  bf16_t* glob_pad = (bf16_t*)take(256);

  bf16_t* pk_emb_n_W0 = takeT(4);
  bf16_t* pk_emb_n_W1 = takeT(8);
  bf16_t* pk_emb_n_W2 = takeT(8);
  bf16_t* pk_emb_e_W0 = takeT(4);
  bf16_t* pk_emb_e_W1 = takeT(8);
  bf16_t* pk_emb_e_W2 = takeT(8);
  bf16_t* pk_dec_W0 = takeT(8);
  bf16_t* pk_dec_W1 = takeT(8);
  bf16_t* pk_dec_W2 = takeT(2);
  bf16_t *pk_eW0[3], *pk_eW1[3], *pk_eW2[3], *pk_nW0[3], *pk_nW1[3], *pk_nW2[3];
  for (int s = 0; s < S; ++s) {
    pk_eW0[s] = takeT(28);
    pk_eW1[s] = takeT(8);
    pk_eW2[s] = takeT(8);
    pk_nW0[s] = takeT(20);
    pk_nW1[s] = takeT(8);
    pk_nW2[s] = takeT(8);
  }

  // --- pack weights into WMMA B-register layout ---
  pack_b_kernel<<<4, 32, 0, stream>>>((const float*)d_in[3], 7, 64, 4, pk_emb_n_W0);
  pack_b_kernel<<<8, 32, 0, stream>>>((const float*)d_in[5], 64, 64, 4, pk_emb_n_W1);
  pack_b_kernel<<<8, 32, 0, stream>>>((const float*)d_in[7], 64, 64, 4, pk_emb_n_W2);
  pack_b_kernel<<<4, 32, 0, stream>>>((const float*)d_in[9], 4, 64, 4, pk_emb_e_W0);
  pack_b_kernel<<<8, 32, 0, stream>>>((const float*)d_in[11], 64, 64, 4, pk_emb_e_W1);
  pack_b_kernel<<<8, 32, 0, stream>>>((const float*)d_in[13], 64, 64, 4, pk_emb_e_W2);
  pack_b_kernel<<<8, 32, 0, stream>>>((const float*)d_in[31], 64, 64, 4, pk_dec_W0);
  pack_b_kernel<<<8, 32, 0, stream>>>((const float*)d_in[33], 64, 64, 4, pk_dec_W1);
  pack_b_kernel<<<2, 32, 0, stream>>>((const float*)d_in[35], 64, 3, 1, pk_dec_W2);
  for (int s = 0; s < S; ++s) {
    pack_b_kernel<<<28, 32, 0, stream>>>((const float*)d_in[15] + (size_t)s * 194 * 64, 194, 64, 4, pk_eW0[s]);
    pack_b_kernel<<<8, 32, 0, stream>>>((const float*)d_in[17] + (size_t)s * 64 * 64, 64, 64, 4, pk_eW1[s]);
    pack_b_kernel<<<8, 32, 0, stream>>>((const float*)d_in[19] + (size_t)s * 64 * 64, 64, 64, 4, pk_eW2[s]);
    pack_b_kernel<<<20, 32, 0, stream>>>((const float*)d_in[21] + (size_t)s * 130 * 64, 130, 64, 4, pk_nW0[s]);
    pack_b_kernel<<<8, 32, 0, stream>>>((const float*)d_in[23] + (size_t)s * 64 * 64, 64, 64, 4, pk_nW1[s]);
    pack_b_kernel<<<8, 32, 0, stream>>>((const float*)d_in[25] + (size_t)s * 64 * 64, 64, 64, 4, pk_nW2[s]);
  }

  // --- pad & convert raw features to bf16 (K padded to 32) ---
  pad_rows_kernel<<<(N * 32 + 255) / 256, 256, 0, stream>>>((const float*)d_in[0], N, 7, 32, Xn);
  pad_rows_kernel<<<(E * 32 + 255) / 256, 256, 0, stream>>>((const float*)d_in[1], E, 4, 32, Xe);
  pad_rows_kernel<<<1, 256, 0, stream>>>(glob, 1, 2, 32, glob_pad);

  const int nodeBlocks = (N / 16 + 3) / 4;  // 782
  const int edgeBlocks = E / 64;            // 12500

  // --- embedders ---
  mlp3_dense_kernel<<<nodeBlocks, 128, 0, stream>>>(
      Xn, 32, 1, pk_emb_n_W0, (const float*)d_in[4], pk_emb_n_W1,
      (const float*)d_in[6], pk_emb_n_W2, (const float*)d_in[8], 4, n_f32, 64,
      64, n_bf16, N);
  mlp3_dense_kernel<<<edgeBlocks, 128, 0, stream>>>(
      Xe, 32, 1, pk_emb_e_W0, (const float*)d_in[10], pk_emb_e_W1,
      (const float*)d_in[12], pk_emb_e_W2, (const float*)d_in[14], 4,
      (float*)nullptr, 64, 64, e_bf16, E);

  // --- message-passing steps ---
  for (int s = 0; s < S; ++s) {
    zero_f32_kernel<<<(N * 64 + 255) / 256, 256, 0, stream>>>(recv, N * 64);
    edge_step_kernel<<<edgeBlocks, 128, 0, stream>>>(
        e_bf16, n_bf16, senders, receivers, glob_pad, pk_eW0[s],
        (const float*)d_in[16] + s * 64, pk_eW1[s],
        (const float*)d_in[18] + s * 64, pk_eW2[s],
        (const float*)d_in[20] + s * 64, new_e, recv, E);
    node_step_kernel<<<nodeBlocks, 128, 0, stream>>>(
        n_bf16, recv, glob_pad, pk_nW0[s], (const float*)d_in[22] + s * 64,
        pk_nW1[s], (const float*)d_in[24] + s * 64, pk_nW2[s],
        (const float*)d_in[26] + s * 64, new_n, N);
    ln_node_kernel<<<(N + 7) / 8, 256, 0, stream>>>(
        n_f32, n_bf16, new_n, (const float*)d_in[27] + s * 64,
        (const float*)d_in[28] + s * 64, N);
    ln_edge_kernel<<<(E + 7) / 8, 256, 0, stream>>>(
        e_bf16, new_e, (const float*)d_in[29] + s * 64,
        (const float*)d_in[30] + s * 64, E);
  }

  // --- decoder: n_bf16 -> [N,3] into d_out ---
  mlp3_dense_kernel<<<nodeBlocks, 128, 0, stream>>>(
      n_bf16, 64, 2, pk_dec_W0, (const float*)d_in[32], pk_dec_W1,
      (const float*)d_in[34], pk_dec_W2, (const float*)d_in[36], 1,
      (float*)d_out, 3, 3, (bf16_t*)nullptr, N);
}
